// SwinBlock_29248727286085
// MI455X (gfx1250) — matrix operand order repaired and checked by reading, compiled, not run on hardware
//
#include <hip/hip_runtime.h>
#include <math.h>

typedef __attribute__((ext_vector_type(16))) _Float16 v16h;
typedef __attribute__((ext_vector_type(8)))  float    v8f;

union Frag16 { v16h h; unsigned int u[8]; };

// ---- problem constants ----
#define IMG_B   64
#define HW      3136        // 56*56
#define CCH     96
#define WSZ     7
#define SHF     3
#define NHEAD   3
#define LL      49
#define SCALEQ  0.17677669529663687f   // 1/sqrt(32)

// packed weight offsets (in halves) inside d_ws
#define QKV_OFF  0           // 3 ktiles * 18 ntiles * 512
#define PROJ_OFF 27648       // 3*6*512
#define MLP1_OFF 36864
#define MLP2_OFF 46080
#define PACK_TOTAL 55296

// ---------------- fragment helpers ----------------
// A fragment (16x32 f16, M x K): lane m=L&15; lanes0-15: K 0..15, lanes16-31: K 16..31
__device__ __forceinline__ void loadA(Frag16& f, const _Float16* base, int row0, int ld, int k0, int lane) {
    int m  = row0 + (lane & 15);
    int hi = (lane >> 4) * 16;
    const char* p = (const char*)(base + m * ld + k0);
#pragma unroll
    for (int pr = 0; pr < 8; ++pr) {
        int j = 2 * pr;
        int k = j + hi;
        f.u[pr] = *(const unsigned int*)(p + 2 * k);
    }
}

// A fragment built from f32 softmax probabilities kept in LDS (ld = 52 floats per row).
// Converts f32 -> f16 on the fly; columns >= 52 are zero (49..51 zeroed by softmax pass).
__device__ __forceinline__ void loadAProb(Frag16& f, const float* S, int row0, int kb, int lane) {
    int m  = row0 + (lane & 15);
    int hi = (lane >> 4) * 16;
    const float* row = S + m * 52;
#pragma unroll
    for (int j = 0; j < 16; ++j) {
        int k = kb + j + hi;
        f.h[j] = (_Float16)((k < 52) ? row[k] : 0.0f);
    }
}

// B fragment from pre-packed register image: lane L holds halves [L*16 .. L*16+15]
__device__ __forceinline__ void loadBP(Frag16& f, const _Float16* p, int lane) {
    const uint4* q = (const uint4*)(p + lane * 16);
    uint4 a = q[0], b = q[1];
    f.u[0] = a.x; f.u[1] = a.y; f.u[2] = a.z; f.u[3] = a.w;
    f.u[4] = b.x; f.u[5] = b.y; f.u[6] = b.z; f.u[7] = b.w;
}

__device__ __forceinline__ v8f wmma32(const Frag16& a, const Frag16& b, v8f c) {
    return __builtin_amdgcn_wmma_f32_16x16x32_f16(false, a.h, false, b.h, (short)0, c, false, false);
}

// ---------------- weight packing: fp32 [K][N] -> f16 B-fragment image ----------------
// B (32x16, KxN): lane n=L&15 holds column n; lanes0-15 K=0..15 (halves j=K), lanes16-31 K=16..31
__global__ void swin_pack_kernel(const float* qkvw, const float* projw,
                                 const float* w1, const float* w2, _Float16* ws) {
    int tid = blockIdx.x * 256 + threadIdx.x;
    if (tid >= PACK_TOTAL) return;
    const float* W; int N, NT, off, elem;
    if (tid < 27648) { W = qkvw; N = 288; NT = 18; off = 0; elem = tid; }
    else {
        int r = tid - 27648; int m = r / 9216; elem = r % 9216;
        N = 96; NT = 6; off = 27648 + m * 9216;
        W = (m == 0) ? projw : (m == 1 ? w1 : w2);
    }
    int tile = elem >> 9;
    int l    = (elem >> 4) & 31;
    int j    = elem & 15;
    int kt = tile / NT, nt = tile % NT;
    int k = kt * 32 + (l >> 4) * 16 + j;
    int n = nt * 16 + (l & 15);
    ws[off + elem] = (_Float16)W[k * N + n];
}

// ---------------- fused windowed attention (LN1 + roll + QKV + attn + proj + residual) ----------------
__global__ __launch_bounds__(256) void swin_attn_kernel(
    const float* __restrict__ x, const _Float16* __restrict__ wpack,
    const float* __restrict__ qkvb, const float* __restrict__ abias,
    const float* __restrict__ projb, const float* __restrict__ ln1g,
    const float* __restrict__ ln1b, float* __restrict__ dout) {

    // LDS layout (62464 B):
    //   Hs   @0      : 64x96 f16 (LN'd window tokens; reused as O f16 after QKV phase)
    //   QKVs @12288  : 64x288 f16
    //   Ss   @49152  : 64 rows * 52 f32 scores; softmax probs written back in-place (f32)
    __shared__ __align__(16) char smem[62464];
    __shared__ int cntA[64];
    _Float16* Hs   = (_Float16*)smem;
    _Float16* QKVs = (_Float16*)(smem + 12288);
    float*    Ss   = (float*)(smem + 49152);

    const int wid = threadIdx.x >> 5, lane = threadIdx.x & 31;
    const int widx = blockIdx.x;
    const int img = widx >> 6, w = widx & 63;
    const int wh = w >> 3, ww = w & 7;

    // shift-mask region ids per window token (computed once)
    if (threadIdx.x < 64) {
        int t = threadIdx.x, c = -1;
        if (t < LL) {
            int gh = wh * 7 + t / 7, gw = ww * 7 + t % 7;
            int ih = gh < 49 ? 0 : (gh < 53 ? 1 : 2);
            int iw = gw < 49 ? 0 : (gw < 53 ? 1 : 2);
            c = ih * 3 + iw;
        }
        cntA[t] = c;
    }

    // ---- load + roll(-3,-3) + LN1 -> Hs (f16), rows 49..63 zero ----
    for (int t = wid; t < 64; t += 8) {
        if (t < LL) {
            int gh = wh * 7 + t / 7, gw = ww * 7 + t % 7;
            int sh = (gh + SHF) % 56, sw = (gw + SHF) % 56;
            const float* src = x + ((size_t)img * HW + sh * 56 + sw) * CCH;
            float v0 = src[lane], v1 = src[lane + 32], v2 = src[lane + 64];
            float s1 = v0 + v1 + v2, s2 = v0 * v0 + v1 * v1 + v2 * v2;
#pragma unroll
            for (int m = 16; m; m >>= 1) { s1 += __shfl_xor(s1, m, 32); s2 += __shfl_xor(s2, m, 32); }
            float mean = s1 * (1.0f / 96.0f);
            float var  = s2 * (1.0f / 96.0f) - mean * mean;
            float rs   = rsqrtf(var + 1e-5f);
            Hs[t * 96 + lane      ] = (_Float16)((v0 - mean) * rs * ln1g[lane]      + ln1b[lane]);
            Hs[t * 96 + lane + 32 ] = (_Float16)((v1 - mean) * rs * ln1g[lane + 32] + ln1b[lane + 32]);
            Hs[t * 96 + lane + 64 ] = (_Float16)((v2 - mean) * rs * ln1g[lane + 64] + ln1b[lane + 64]);
        } else {
            Hs[t * 96 + lane] = (_Float16)0.f;
            Hs[t * 96 + lane + 32] = (_Float16)0.f;
            Hs[t * 96 + lane + 64] = (_Float16)0.f;
        }
    }
    __syncthreads();

    // ---- QKV GEMM: [64x96] @ [96x288], scale folded into Q ----
    for (int idx = wid; idx < 72; idx += 8) {
        int mt = idx / 18, nt = idx % 18;
        v8f c = {};
#pragma unroll
        for (int ks = 0; ks < 3; ++ks) {
            Frag16 a, b;
            loadA(a, Hs, mt * 16, 96, ks * 32, lane);
            loadBP(b, wpack + QKV_OFF + (ks * 18 + nt) * 512, lane);
            c = wmma32(a, b, c);
        }
        int n = lane & 15, hi = lane >> 4, col = nt * 16 + n;
        float bb = qkvb[col];
        float sc = (col < 96) ? SCALEQ : 1.0f;
#pragma unroll
        for (int i = 0; i < 8; ++i) {
            int row = mt * 16 + i + hi * 8;
            QKVs[row * 288 + col] = (_Float16)((c[i] + bb) * sc);
        }
    }
    __syncthreads();

    // ---- attention per head ----
    for (int hh = 0; hh < NHEAD; ++hh) {
        // scores = Q @ K^T + bias + mask
        for (int idx = wid; idx < 16; idx += 8) {
            int mt = idx >> 2, nt = idx & 3;
            Frag16 a, b;
            loadA(a, QKVs, mt * 16, 288, hh * 32, lane);  // Q slice
            {   // K^T fragment gathered from LDS (column = key token)
                int n = lane & 15, hi = lane >> 4;
                int tk = nt * 16 + n;
                const char* p = (const char*)(QKVs + tk * 288 + 96 + hh * 32);
#pragma unroll
                for (int pr = 0; pr < 8; ++pr) {
                    int k = hi * 16 + 2 * pr;
                    b.u[pr] = *(const unsigned int*)(p + 2 * k);
                }
            }
            v8f c = {};
            c = wmma32(a, b, c);
            int n = lane & 15, hi = lane >> 4, kk = nt * 16 + n;
#pragma unroll
            for (int i = 0; i < 8; ++i) {
                int q = mt * 16 + i + hi * 8;
                float v;
                if (q < LL && kk < LL) {
                    float msk = (cntA[q] != cntA[kk]) ? -100.0f : 0.0f;
                    v = c[i] + abias[hh * 2401 + q * 49 + kk] + msk;
                } else v = -1e30f;
                if (kk < 52) Ss[q * 52 + kk] = v;
            }
        }
        __syncthreads();
        // softmax (row per thread), probs written back in-place as f32 (same type, no aliasing)
        if (threadIdx.x < 64) {
            int t = threadIdx.x;
            float* Sr = Ss + t * 52;
            float mx = -1e30f;
            for (int c2 = 0; c2 < LL; ++c2) mx = fmaxf(mx, Sr[c2]);
            float sum = 0.f;
            for (int c2 = 0; c2 < LL; ++c2) sum += __expf(Sr[c2] - mx);
            float inv = 1.0f / sum;
            for (int c2 = 0; c2 < LL; ++c2) Sr[c2] = __expf(Sr[c2] - mx) * inv;
            Sr[49] = 0.f; Sr[50] = 0.f; Sr[51] = 0.f;
        }
        __syncthreads();
        // O = P @ V  (K dim = 64 keys, 2 steps); P converted f32->f16 in the loader
        for (int idx = wid; idx < 8; idx += 8) {
            int mt = idx >> 1, nd = idx & 1;
            v8f c = {};
#pragma unroll
            for (int kb = 0; kb < 64; kb += 32) {
                Frag16 a, b;
                loadAProb(a, Ss, mt * 16, kb, lane);
                {   // V fragment: column = head dim, k = key token (strided gather)
                    int n = lane & 15, hi = lane >> 4;
                    int col = 192 + hh * 32 + nd * 16 + n;
#pragma unroll
                    for (int j = 0; j < 16; ++j) {
                        int k = kb + hi * 16 + j;
                        b.h[j] = QKVs[k * 288 + col];
                    }
                }
                c = wmma32(a, b, c);
            }
            int n = lane & 15, hi = lane >> 4;
            int dcol = hh * 32 + nd * 16 + n;
#pragma unroll
            for (int i = 0; i < 8; ++i) {
                int row = mt * 16 + i + hi * 8;
                Hs[row * 96 + dcol] = (_Float16)c[i];   // O f16 reuses Hs
            }
        }
        __syncthreads();
    }

    // ---- proj GEMM + reverse-roll scatter + shortcut residual ----
    for (int idx = wid; idx < 24; idx += 8) {
        int mt = idx / 6, nt = idx % 6;
        v8f c = {};
#pragma unroll
        for (int ks = 0; ks < 3; ++ks) {
            Frag16 a, b;
            loadA(a, Hs, mt * 16, 96, ks * 32, lane);
            loadBP(b, wpack + PROJ_OFF + (ks * 6 + nt) * 512, lane);
            c = wmma32(a, b, c);
        }
        int n = lane & 15, hi = lane >> 4, cc = nt * 16 + n;
        float pb = projb[cc];
#pragma unroll
        for (int i = 0; i < 8; ++i) {
            int t = mt * 16 + i + hi * 8;
            if (t < LL) {
                int gh = wh * 7 + t / 7, gw = ww * 7 + t % 7;
                int fh = (gh + SHF) % 56, fw = (gw + SHF) % 56;
                size_t o = ((size_t)img * HW + fh * 56 + fw) * CCH + cc;
                dout[o] = c[i] + pb + x[o];
            }
        }
    }
}

// ---------------- fused LN2 + MLP (GEMM, exact GELU, GEMM) + residual, in-place on dout ----------------
__global__ __launch_bounds__(256) void swin_mlp_kernel(
    const _Float16* __restrict__ wpack, const float* __restrict__ b1,
    const float* __restrict__ b2, const float* __restrict__ g2,
    const float* __restrict__ be2, float* __restrict__ dout) {

    // LDS: X2 f32 @0 (24576) | Xh f16 @24576 (12288) | Yh f16 @36864 (12288)
    __shared__ __align__(16) char smem[49152];
    float*    X2 = (float*)smem;
    _Float16* Xh = (_Float16*)(smem + 24576);
    _Float16* Yh = (_Float16*)(smem + 36864);

    const int wid = threadIdx.x >> 5, lane = threadIdx.x & 31;
    const size_t base = (size_t)blockIdx.x * 64 * CCH;

    // load 64 tokens of x2
    for (int i = threadIdx.x; i < 1536; i += 256)
        ((float4*)X2)[i] = ((const float4*)(dout + base))[i];
    __syncthreads();

    // LN2 per row
    for (int t = wid; t < 64; t += 8) {
        float v0 = X2[t * 96 + lane], v1 = X2[t * 96 + lane + 32], v2 = X2[t * 96 + lane + 64];
        float s1 = v0 + v1 + v2, s2 = v0 * v0 + v1 * v1 + v2 * v2;
#pragma unroll
        for (int m = 16; m; m >>= 1) { s1 += __shfl_xor(s1, m, 32); s2 += __shfl_xor(s2, m, 32); }
        float mean = s1 * (1.0f / 96.0f);
        float var  = s2 * (1.0f / 96.0f) - mean * mean;
        float rs   = rsqrtf(var + 1e-5f);
        Xh[t * 96 + lane      ] = (_Float16)((v0 - mean) * rs * g2[lane]      + be2[lane]);
        Xh[t * 96 + lane + 32 ] = (_Float16)((v1 - mean) * rs * g2[lane + 32] + be2[lane + 32]);
        Xh[t * 96 + lane + 64 ] = (_Float16)((v2 - mean) * rs * g2[lane + 64] + be2[lane + 64]);
    }
    __syncthreads();

    // GEMM1 + exact GELU
    for (int idx = wid; idx < 24; idx += 8) {
        int mt = idx / 6, nt = idx % 6;
        v8f c = {};
#pragma unroll
        for (int ks = 0; ks < 3; ++ks) {
            Frag16 a, b;
            loadA(a, Xh, mt * 16, 96, ks * 32, lane);
            loadBP(b, wpack + MLP1_OFF + (ks * 6 + nt) * 512, lane);
            c = wmma32(a, b, c);
        }
        int n = lane & 15, hi = lane >> 4, cc = nt * 16 + n;
        float bb = b1[cc];
#pragma unroll
        for (int i = 0; i < 8; ++i) {
            int row = mt * 16 + i + hi * 8;
            float v = c[i] + bb;
            float g = 0.5f * v * (1.0f + erff(v * 0.7071067811865476f));
            Yh[row * 96 + cc] = (_Float16)g;
        }
    }
    __syncthreads();

    // GEMM2 + residual
    for (int idx = wid; idx < 24; idx += 8) {
        int mt = idx / 6, nt = idx % 6;
        v8f c = {};
#pragma unroll
        for (int ks = 0; ks < 3; ++ks) {
            Frag16 a, b;
            loadA(a, Yh, mt * 16, 96, ks * 32, lane);
            loadBP(b, wpack + MLP2_OFF + (ks * 6 + nt) * 512, lane);
            c = wmma32(a, b, c);
        }
        int n = lane & 15, hi = lane >> 4, cc = nt * 16 + n;
        float bb = b2[cc];
#pragma unroll
        for (int i = 0; i < 8; ++i) {
            int row = mt * 16 + i + hi * 8;
            dout[base + row * 96 + cc] = c[i] + bb + X2[row * 96 + cc];
        }
    }
}

extern "C" void kernel_launch(void* const* d_in, const int* in_sizes, int n_in,
                              void* d_out, int out_size, void* d_ws, size_t ws_size,
                              hipStream_t stream) {
    const float* x      = (const float*)d_in[0];
    const float* qkv_w  = (const float*)d_in[1];
    const float* qkv_b  = (const float*)d_in[2];
    const float* abias  = (const float*)d_in[3];
    const float* proj_w = (const float*)d_in[4];
    const float* proj_b = (const float*)d_in[5];
    const float* ln1_g  = (const float*)d_in[6];
    const float* ln1_b  = (const float*)d_in[7];
    const float* ln2_g  = (const float*)d_in[8];
    const float* ln2_b  = (const float*)d_in[9];
    const float* mlp_w1 = (const float*)d_in[10];
    const float* mlp_b1 = (const float*)d_in[11];
    const float* mlp_w2 = (const float*)d_in[12];
    const float* mlp_b2 = (const float*)d_in[13];
    float* out = (float*)d_out;
    _Float16* wpack = (_Float16*)d_ws;

    (void)in_sizes; (void)n_in; (void)out_size; (void)ws_size;

    swin_pack_kernel<<<216, 256, 0, stream>>>(qkv_w, proj_w, mlp_w1, mlp_w2, wpack);
    swin_attn_kernel<<<IMG_B * 64, 256, 0, stream>>>(x, wpack, qkv_b, abias, proj_b,
                                                     ln1_g, ln1_b, out);
    swin_mlp_kernel<<<HW * IMG_B / 64, 256, 0, stream>>>(wpack, mlp_b1, mlp_b2,
                                                         ln2_g, ln2_b, out);
}
